// SO2EquivariantGraphAttention_18588618457697
// MI455X (gfx1250) — matrix operand order repaired and checked by reading, not compiled, not run on hardware
//
#include <hip/hip_runtime.h>
#include <hip/hip_bf16.h>
#include <math.h>

// ---------------------------------------------------------------------------
// SO2 equivariant graph attention, MI455X (gfx1250).
// All dense GEMMs on v_wmma_f32_16x16x32_bf16 (bf16 A/B, f32 accum).
// Weights pre-packed (and SO2 complex-fold pre-applied) to bf16 once per
// launch; A-tiles staged in LDS as bf16 so the WMMA inner loops are pure
// ds_load_b128 / global_load_b128 + v_wmma with no per-fragment conversions.
// ---------------------------------------------------------------------------

typedef __bf16 bf16;
typedef __attribute__((ext_vector_type(16))) __bf16 v16bf;
typedef __attribute__((ext_vector_type(8)))  __bf16 v8bf;
typedef __attribute__((ext_vector_type(8)))  float  v8f;

constexpr int kE = 60000;   // edges
constexpr int kN = 15000;   // nodes

// ---------------------------------------------------------------------------
// WMMA helpers (wave32, 16x16x32 bf16)
// ---------------------------------------------------------------------------
__device__ __forceinline__ v8f wmma_bf16(v16bf a, v16bf b, v8f c) {
  return __builtin_amdgcn_wmma_f32_16x16x32_bf16(
      /*neg_a=*/false, a, /*neg_b=*/false, b,
      /*c_mod=*/(short)0, c, /*reuse_a=*/false, /*reuse_b=*/false);
}

// A fragment from bf16 LDS tile (16 x lda, row-major). Lane layout (contiguous k, same as frag_b):
// lanes 0-15 row m=lane, K = k0+0..15; lanes 16-31 K = k0+16..31.
__device__ __forceinline__ v16bf frag_a_ldsbf(const bf16* A, int lda, int k0) {
  const int lane = threadIdx.x & 31;
  const bf16* p = A + (lane & 15) * lda + k0 + ((lane >> 4) << 4);
  const v8bf lo = *reinterpret_cast<const v8bf*>(p);
  const v8bf hi = *reinterpret_cast<const v8bf*>(p + 8);
  v16bf a;
#pragma unroll
  for (int j = 0; j < 8; ++j) { a[j] = lo[j]; a[8 + j] = hi[j]; }
  return a;
}

// B fragment: B[k][n] = W[n0+n][k0+k], W bf16 row-major (rows=outputs, ldk).
// Per lane the 16 K-values are contiguous -> one 32B vector load.
__device__ __forceinline__ v16bf frag_b_bf16(const bf16* W, int ldk, int n0, int k0) {
  const int lane = threadIdx.x & 31;
  const bf16* p = W + (size_t)(n0 + (lane & 15)) * ldk + k0 + ((lane >> 4) << 4);
  return *reinterpret_cast<const v16bf*>(p);
}

// Generic 16x16 output tile: C = A(16 x 32*ksteps, LDS bf16) x W^T tile
__device__ __forceinline__ v8f gemm_tile(const bf16* As, int lda, const bf16* Wb,
                                         int ldk, int nt, int ksteps) {
  v8f c = {};
  for (int kb = 0; kb < ksteps; ++kb)
    c = wmma_bf16(frag_a_ldsbf(As, lda, kb * 32), frag_b_bf16(Wb, ldk, nt * 16, kb * 32), c);
  return c;
}

__device__ __forceinline__ void pack_store(bf16* p, float a, float b) {
  union { bf16 h[2]; unsigned u; } pk;
  pk.h[0] = (bf16)a; pk.h[1] = (bf16)b;
  *reinterpret_cast<unsigned*>(p) = pk.u;   // even element index -> 4B aligned
}

__device__ __forceinline__ void ln_silu_row_bf(const float* row, bf16* out,
                                               const float* g, const float* b) {
  float mu = 0.f;
#pragma unroll
  for (int i = 0; i < 64; ++i) mu += row[i];
  mu *= (1.f / 64.f);
  float var = 0.f;
#pragma unroll
  for (int i = 0; i < 64; ++i) { float d = row[i] - mu; var += d * d; }
  var *= (1.f / 64.f);
  const float rs = rsqrtf(var + 1e-5f);
#pragma unroll
  for (int i = 0; i < 64; i += 2) {
    float v0 = (row[i]     - mu) * rs * g[i]     + b[i];
    float v1 = (row[i + 1] - mu) * rs * g[i + 1] + b[i + 1];
    v0 = v0 / (1.f + __expf(-v0));
    v1 = v1 / (1.f + __expf(-v1));
    pack_store(out + i, v0, v1);
  }
}

__device__ __forceinline__ void atomicMaxFloat(float* addr, float val) {
  if (val >= 0.f) atomicMax((int*)addr, __float_as_int(val));
  else            atomicMin((unsigned int*)addr, (unsigned int)__float_as_int(val));
}

// ---------------------------------------------------------------------------
// Prep kernels: bf16 transcode + SO2 complex-fold of m-block weights.
// ---------------------------------------------------------------------------
__global__ void k_cvt(const float* __restrict__ s, bf16* __restrict__ d, int n) {
  const int i = blockIdx.x * blockDim.x + threadIdx.x;
  if (i < n) d[i] = (bf16)s[i];
}

// Wm is (2h x K) fp32. Build Wf (2h x 2K) bf16 with
//   out0 (n'<h):  k'<K -> +Wm[n'][k'],   k'>=K -> -Wm[h+n'][k'-K]
//   out1 (n'>=h): k'<K -> +Wm[h+nl][k'], k'>=K -> +Wm[nl][k'-K]
__global__ void k_fold(const float* __restrict__ Wm, bf16* __restrict__ Wf,
                       int K, int h, int n) {
  const int i = blockIdx.x * blockDim.x + threadIdx.x;
  if (i >= n) return;
  const int K2 = 2 * K;
  const int np = i / K2, kp = i % K2;
  const int nh = np >= h, nl = np - (nh ? h : 0);
  const int kh = kp >= K, kl = kp - (kh ? K : 0);
  float v;
  if (!nh && !kh)      v =  Wm[(size_t)nl * K + kl];
  else if (!nh &&  kh) v = -Wm[(size_t)(h + nl) * K + kl];
  else if ( nh && !kh) v =  Wm[(size_t)(h + nl) * K + kl];
  else                 v =  Wm[(size_t)nl * K + kl];
  Wf[i] = (bf16)v;
}

__global__ void k_init(float* amax, float* denom, int n) {
  const int i = blockIdx.x * blockDim.x + threadIdx.x;
  if (i < n) { amax[i] = -INFINITY; denom[i] = 0.f; }
}

// ---------------------------------------------------------------------------
// Kernel 1: radial MLP: x_edge(E,256) -> LN/silu 64 -> LN/silu 64 -> 768
// one block = 16 edges, 4 waves, fused WMMA GEMMs
// ---------------------------------------------------------------------------
__launch_bounds__(128)
__global__ void k_rad_mlp(const float* __restrict__ edge_distance,
                          const int*   __restrict__ atomic_numbers,
                          const int*   __restrict__ edge_index,
                          const float* __restrict__ src_emb,
                          const float* __restrict__ tgt_emb,
                          const bf16* __restrict__ w1b, const float* __restrict__ b1,
                          const float* __restrict__ g1, const float* __restrict__ bb1,
                          const bf16* __restrict__ w2b, const float* __restrict__ b2,
                          const float* __restrict__ g2, const float* __restrict__ bb2,
                          const bf16* __restrict__ w3b, const float* __restrict__ b3,
                          float* __restrict__ rad) {
  __shared__ __align__(32) bf16 xtb[16 * 256];
  __shared__ float yt[16 * 64];
  __shared__ __align__(32) bf16 ytb[16 * 64];
  __shared__ float zt[16 * 64];
  __shared__ __align__(32) bf16 ztb[16 * 64];
  const int e0   = blockIdx.x * 16;
  const int wave = threadIdx.x >> 5;
  const int lane = threadIdx.x & 31;

  // stage x_edge = [edge_distance | src_emb | tgt_emb] as bf16 pairs
  for (int i = threadIdx.x; i < 16 * 128; i += 128) {
    const int el = i >> 7, c = (i & 127) * 2;
    const int e = e0 + el;
    float v0, v1;
    if (c < 128) {
      v0 = edge_distance[(size_t)e * 128 + c];
      v1 = edge_distance[(size_t)e * 128 + c + 1];
    } else if (c < 192) {
      const float* p = src_emb + atomic_numbers[edge_index[e]] * 64 + (c - 128);
      v0 = p[0]; v1 = p[1];
    } else {
      const float* p = tgt_emb + atomic_numbers[edge_index[kE + e]] * 64 + (c - 192);
      v0 = p[0]; v1 = p[1];
    }
    pack_store(&xtb[el * 256 + c], v0, v1);
  }
  __syncthreads();

  { // layer 1: 16x256 @ 256x64
    const int nt = wave;
    v8f c = gemm_tile(xtb, 256, w1b, 256, nt, 8);
    const int col = nt * 16 + (lane & 15);
    const float bias = b1[col];
#pragma unroll
    for (int r = 0; r < 8; ++r) yt[(r + ((lane >> 4) << 3)) * 64 + col] = c[r] + bias;
  }
  __syncthreads();
  if (threadIdx.x < 16) ln_silu_row_bf(&yt[threadIdx.x * 64], &ytb[threadIdx.x * 64], g1, bb1);
  __syncthreads();

  { // layer 2: 16x64 @ 64x64
    const int nt = wave;
    v8f c = gemm_tile(ytb, 64, w2b, 64, nt, 2);
    const int col = nt * 16 + (lane & 15);
    const float bias = b2[col];
#pragma unroll
    for (int r = 0; r < 8; ++r) zt[(r + ((lane >> 4) << 3)) * 64 + col] = c[r] + bias;
  }
  __syncthreads();
  if (threadIdx.x < 16) ln_silu_row_bf(&zt[threadIdx.x * 64], &ztb[threadIdx.x * 64], g2, bb2);
  __syncthreads();

  // layer 3: 16x64 @ 64x768
  for (int nt = wave; nt < 48; nt += 4) {
    v8f c = gemm_tile(ztb, 64, w3b, 64, nt, 2);
    const int col = nt * 16 + (lane & 15);
    const float bias = b3[col];
#pragma unroll
    for (int r = 0; r < 8; ++r) {
      const int row = r + ((lane >> 4) << 3);
      rad[(size_t)(e0 + row) * 768 + col] = c[r] + bias;
    }
  }
}

// ---------------------------------------------------------------------------
// Kernel 2: msgW = wigner(9x9) @ [x[src]|x[dst]](9x128), one wave per edge
// ---------------------------------------------------------------------------
__launch_bounds__(256)
__global__ void k_wigner_msg(const float* __restrict__ x,
                             const float* __restrict__ wig,
                             const int*   __restrict__ edge_index,
                             float* __restrict__ msgW) {
  const int e    = blockIdx.x * 8 + (threadIdx.x >> 5);
  const int lane = threadIdx.x & 31;
  const int src = edge_index[e];
  const int dst = edge_index[kE + e];

  v16bf a;
  {
    const int m  = lane & 15;
    const int kb = (lane >> 4) << 4;
    const float* wp = wig + (size_t)e * 81 + m * 9;
#pragma unroll
    for (int j = 0; j < 8; ++j) {
      const int k1 = kb + j, k2 = kb + 8 + j;
      a[j]     = (bf16)((m < 9 && k1 < 9) ? wp[k1] : 0.f);
      a[8 + j] = (bf16)((m < 9 && k2 < 9) ? wp[k2] : 0.f);
    }
  }
  const float* xs = x + (size_t)src * 576;
  const float* xd = x + (size_t)dst * 576;
  for (int nt = 0; nt < 8; ++nt) {
    const int n = nt * 16 + (lane & 15);
    const float* xp = (n < 64) ? (xs + n) : (xd + (n - 64));
    const int kb = (lane >> 4) << 4;
    v16bf b;
#pragma unroll
    for (int j = 0; j < 16; ++j) {
      const int k = kb + j;
      b[j] = (bf16)((k < 9) ? xp[(size_t)k * 64] : 0.f);
    }
    v8f c = {};
    c = wmma_bf16(a, b, c);
#pragma unroll
    for (int r = 0; r < 8; ++r) {
      const int row = r + ((lane >> 4) << 3);
      if (row < 9) msgW[(size_t)e * 1152 + row * 128 + n] = c[r];
    }
  }
}

// ---------------------------------------------------------------------------
// Kernel 3: SO2 conv #1 (cin=128, cout=64, rad-scaled, extra=576)
// block = 16 edges, 8 waves; concat row j -> output row M_PERM[j].
// ---------------------------------------------------------------------------
__launch_bounds__(256)
__global__ void k_conv1(const float* __restrict__ msgW, const float* __restrict__ rad,
                        const bf16* __restrict__ w0b, const float* __restrict__ b0,
                        const bf16* __restrict__ w1f, const bf16* __restrict__ w2f,
                        float* __restrict__ extraD, float* __restrict__ msg1) {
  __shared__ __align__(32) bf16 As[16 * 512];
  const int e0   = blockIdx.x * 16;
  const int wave = threadIdx.x >> 5;
  const int lane = threadIdx.x & 31;

  // ---- m=0: A = rad-scaled rows {0,2,6}, K=384 -> N=768 ----
  for (int i = threadIdx.x; i < 16 * 192; i += 256) {
    const int m = i / 192, k = (i % 192) * 2;
    const int e = e0 + m;
    const int l = k >> 7, c = k & 127;
    const int srow = (l == 0) ? 0 : (l == 1 ? 2 : 6);
    const float* mp = msgW + (size_t)e * 1152 + srow * 128 + c;
    const float* rp = rad + (size_t)e * 768 + k;
    pack_store(&As[m * 384 + k], mp[0] * rp[0], mp[1] * rp[1]);
  }
  __syncthreads();
  for (int nt = wave; nt < 48; nt += 8) {
    v8f c = gemm_tile(As, 384, w0b, 384, nt, 12);
    const int col = nt * 16 + (lane & 15);
    const float bias = b0[col];
#pragma unroll
    for (int r = 0; r < 8; ++r) {
      const int row = r + ((lane >> 4) << 3);
      const int e = e0 + row;
      const float v = c[r] + bias;
      if (col < 576) {
        extraD[(size_t)e * 576 + col] = v;               // alpha(512)+gating(64)
      } else {
        const int cp = col - 576;
        const int l = cp >> 6, cc = cp & 63;
        const int orow = (l == 0) ? 0 : (l == 1 ? 2 : 6);
        msg1[(size_t)e * 576 + orow * 64 + cc] = v;
      }
    }
  }
  __syncthreads();

  // ---- m=1: A' = [xb0|xb1] (16x512), folded Wf (256x512) ----
  for (int i = threadIdx.x; i < 16 * 256; i += 256) {
    const int m = i >> 8, k = (i & 255) * 2;
    const int e = e0 + m;
    const int comp = k >> 8, kk = k & 255;
    const int srow = (kk < 128) ? (comp ? 3 : 1) : (comp ? 7 : 5);
    const float* mp = msgW + (size_t)e * 1152 + srow * 128 + (kk & 127);
    const float* rp = rad + (size_t)e * 768 + 384 + kk;
    pack_store(&As[m * 512 + k], mp[0] * rp[0], mp[1] * rp[1]);
  }
  __syncthreads();
  for (int nt = wave; nt < 16; nt += 8) {
    v8f c = gemm_tile(As, 512, w1f, 512, nt, 16);
    const int ng = nt * 16 + (lane & 15);
    const int nh = ng >= 128, nl = ng & 127;
    const int row9 = nh ? (nl < 64 ? 3 : 7) : (nl < 64 ? 1 : 5);
    const int cc = nl & 63;
#pragma unroll
    for (int r = 0; r < 8; ++r) {
      const int row = r + ((lane >> 4) << 3);
      msg1[(size_t)(e0 + row) * 576 + row9 * 64 + cc] = c[r];
    }
  }
  __syncthreads();

  // ---- m=2: A' (16x256), folded Wf (128x256) ----
  for (int i = threadIdx.x; i < 16 * 128; i += 256) {
    const int m = i >> 7, k = (i & 127) * 2;
    const int e = e0 + m;
    const int comp = k >> 7, kk = k & 127;
    const int srow = comp ? 8 : 4;
    const float* mp = msgW + (size_t)e * 1152 + srow * 128 + kk;
    const float* rp = rad + (size_t)e * 768 + 640 + kk;
    pack_store(&As[m * 256 + k], mp[0] * rp[0], mp[1] * rp[1]);
  }
  __syncthreads();
  for (int nt = wave; nt < 8; nt += 8) {
    v8f c = gemm_tile(As, 256, w2f, 256, nt, 8);
    const int ng = nt * 16 + (lane & 15);
    const int nh = ng >= 64, nl = ng & 63;
    const int row9 = nh ? 8 : 4;
#pragma unroll
    for (int r = 0; r < 8; ++r) {
      const int row = r + ((lane >> 4) << 3);
      msg1[(size_t)(e0 + row) * 576 + row9 * 64 + nl] = c[r];
    }
  }
}

// ---------------------------------------------------------------------------
// Kernel 4: attention logits: LN(head) -> smooth_leaky -> dot -> atomic max
// ---------------------------------------------------------------------------
__global__ void k_logits(const float* __restrict__ extraD,
                         const float* __restrict__ aln_g, const float* __restrict__ aln_b,
                         const float* __restrict__ alpha_dot,
                         const int* __restrict__ edge_index,
                         float* __restrict__ logits, float* __restrict__ amax) {
  const int i = blockIdx.x * blockDim.x + threadIdx.x;
  if (i >= kE * 8) return;
  const int e = i >> 3, h = i & 7;
  const float* a = extraD + (size_t)e * 576 + h * 64;
  float mu = 0.f;
#pragma unroll
  for (int j = 0; j < 64; ++j) mu += a[j];
  mu *= (1.f / 64.f);
  float var = 0.f;
#pragma unroll
  for (int j = 0; j < 64; ++j) { const float d = a[j] - mu; var += d * d; }
  var *= (1.f / 64.f);
  const float rs = rsqrtf(var + 1e-5f);
  float s = 0.f;
#pragma unroll
  for (int j = 0; j < 64; ++j) {
    float v = (a[j] - mu) * rs * aln_g[j] + aln_b[j];
    const float sig = 1.f / (1.f + __expf(-v));
    const float sl = 0.6f * v + 0.4f * v * (2.f * sig - 1.f);  // smooth_leaky a=0.2
    s += sl * alpha_dot[h * 64 + j];
  }
  logits[i] = s;
  atomicMaxFloat(&amax[edge_index[kE + e] * 8 + h], s);
}

// Kernel 5: ea = exp(logit - amax[dst]); denom += ea
__global__ void k_ea(const float* __restrict__ amax, const int* __restrict__ edge_index,
                     float* __restrict__ logits, float* __restrict__ denom) {
  const int i = blockIdx.x * blockDim.x + threadIdx.x;
  if (i >= kE * 8) return;
  const int e = i >> 3, h = i & 7;
  const int d = edge_index[kE + e];
  const float ea = __expf(logits[i] - amax[d * 8 + h]);
  logits[i] = ea;
  atomicAdd(&denom[d * 8 + h], ea);
}

// ---------------------------------------------------------------------------
// Kernel 6: S2 grid act: s2 = from_grid @ silu(to_grid @ msg); row0 = silu(gate)
// ---------------------------------------------------------------------------
__global__ void k_grid(const float* __restrict__ to_grid, const float* __restrict__ from_grid,
                       const float* __restrict__ extraD, float* __restrict__ msg1) {
  const int i = blockIdx.x * blockDim.x + threadIdx.x;
  if (i >= kE * 64) return;
  const int e = i >> 6, c = i & 63;
  float m[9];
#pragma unroll
  for (int k = 0; k < 9; ++k) m[k] = msg1[(size_t)e * 576 + k * 64 + c];
  float s2[9];
#pragma unroll
  for (int k = 0; k < 9; ++k) s2[k] = 0.f;
  for (int g = 0; g < 16; ++g) {
    float xv = 0.f;
#pragma unroll
    for (int k = 0; k < 9; ++k) xv += to_grid[g * 9 + k] * m[k];
    xv = xv / (1.f + __expf(-xv));
#pragma unroll
    for (int k = 0; k < 9; ++k) s2[k] += from_grid[k * 16 + g] * xv;
  }
  float gate = extraD[(size_t)e * 576 + 512 + c];
  gate = gate / (1.f + __expf(-gate));
  msg1[(size_t)e * 576 + c] = gate;
#pragma unroll
  for (int k = 1; k < 9; ++k) msg1[(size_t)e * 576 + k * 64 + c] = s2[k];
}

// ---------------------------------------------------------------------------
// Kernel 7: SO2 conv #2 (cin=64, cout=128)
// ---------------------------------------------------------------------------
__launch_bounds__(256)
__global__ void k_conv2(const float* __restrict__ msg1,
                        const bf16* __restrict__ w0b, const float* __restrict__ b0,
                        const bf16* __restrict__ w1f, const bf16* __restrict__ w2f,
                        float* __restrict__ msg2) {
  __shared__ __align__(32) bf16 As[16 * 256];
  const int e0   = blockIdx.x * 16;
  const int wave = threadIdx.x >> 5;
  const int lane = threadIdx.x & 31;

  // m=0: rows {0,2,6}, K=192 -> N=384
  for (int i = threadIdx.x; i < 16 * 96; i += 256) {
    const int m = i / 96, k = (i % 96) * 2;
    const int l = k >> 6, c = k & 63;
    const int srow = (l == 0) ? 0 : (l == 1 ? 2 : 6);
    const float* mp = msg1 + (size_t)(e0 + m) * 576 + srow * 64 + c;
    pack_store(&As[m * 192 + k], mp[0], mp[1]);
  }
  __syncthreads();
  for (int nt = wave; nt < 24; nt += 8) {
    v8f c = gemm_tile(As, 192, w0b, 192, nt, 6);
    const int col = nt * 16 + (lane & 15);
    const float bias = b0[col];
    const int l = col >> 7, cc = col & 127;
    const int orow = (l == 0) ? 0 : (l == 1 ? 2 : 6);
#pragma unroll
    for (int r = 0; r < 8; ++r) {
      const int row = r + ((lane >> 4) << 3);
      msg2[(size_t)(e0 + row) * 1152 + orow * 128 + cc] = c[r] + bias;
    }
  }
  __syncthreads();

  // m=1: A' (16x256), folded Wf (512x256)
  for (int i = threadIdx.x; i < 16 * 128; i += 256) {
    const int m = i >> 7, k = (i & 127) * 2;
    const int comp = k >> 7, kk = k & 127;
    const int srow = (kk < 64) ? (comp ? 3 : 1) : (comp ? 7 : 5);
    const float* mp = msg1 + (size_t)(e0 + m) * 576 + srow * 64 + (kk & 63);
    pack_store(&As[m * 256 + k], mp[0], mp[1]);
  }
  __syncthreads();
  for (int nt = wave; nt < 32; nt += 8) {
    v8f c = gemm_tile(As, 256, w1f, 256, nt, 8);
    const int ng = nt * 16 + (lane & 15);
    const int nh = ng >= 256, nl = ng & 255;
    const int row9 = nh ? (nl < 128 ? 3 : 7) : (nl < 128 ? 1 : 5);
    const int cc = nl & 127;
#pragma unroll
    for (int r = 0; r < 8; ++r) {
      const int row = r + ((lane >> 4) << 3);
      msg2[(size_t)(e0 + row) * 1152 + row9 * 128 + cc] = c[r];
    }
  }
  __syncthreads();

  // m=2: A' (16x128), folded Wf (256x128)
  for (int i = threadIdx.x; i < 16 * 64; i += 256) {
    const int m = i >> 6, k = (i & 63) * 2;
    const int comp = k >> 6, kk = k & 63;
    const int srow = comp ? 8 : 4;
    const float* mp = msg1 + (size_t)(e0 + m) * 576 + srow * 64 + kk;
    pack_store(&As[m * 128 + k], mp[0], mp[1]);
  }
  __syncthreads();
  for (int nt = wave; nt < 16; nt += 8) {
    v8f c = gemm_tile(As, 128, w2f, 128, nt, 4);
    const int ng = nt * 16 + (lane & 15);
    const int nh = ng >= 128, nl = ng & 127;
    const int row9 = nh ? 8 : 4;
#pragma unroll
    for (int r = 0; r < 8; ++r) {
      const int row = r + ((lane >> 4) << 3);
      msg2[(size_t)(e0 + row) * 1152 + row9 * 128 + nl] = c[r];
    }
  }
}

// ---------------------------------------------------------------------------
// Kernel 8: attn = wigner_inv @ (msg2 * alpha), atomic segment-sum into node
// ---------------------------------------------------------------------------
__launch_bounds__(256)
__global__ void k_attn_agg(const float* __restrict__ msg2,
                           const float* __restrict__ wig_inv,
                           const int*   __restrict__ edge_index,
                           const float* __restrict__ ea, const float* __restrict__ denom,
                           float* __restrict__ node) {
  const int e    = blockIdx.x * 8 + (threadIdx.x >> 5);
  const int lane = threadIdx.x & 31;
  const int d = edge_index[kE + e];

  v16bf a;
  {
    const int m  = lane & 15;
    const int kb = (lane >> 4) << 4;
    const float* wp = wig_inv + (size_t)e * 81 + m * 9;
#pragma unroll
    for (int j = 0; j < 8; ++j) {
      const int k1 = kb + j, k2 = kb + 8 + j;
      a[j]     = (bf16)((m < 9 && k1 < 9) ? wp[k1] : 0.f);
      a[8 + j] = (bf16)((m < 9 && k2 < 9) ? wp[k2] : 0.f);
    }
  }
  for (int nt = 0; nt < 8; ++nt) {             // nt == head (V==16)
    const float alpha = ea[e * 8 + nt] / (denom[d * 8 + nt] + 1e-16f);
    const int n = nt * 16 + (lane & 15);
    const float* mp = msg2 + (size_t)e * 1152 + n;
    const int kb = (lane >> 4) << 4;
    v16bf b;
#pragma unroll
    for (int j = 0; j < 16; ++j) {
      const int k = kb + j;
      b[j] = (bf16)((k < 9) ? mp[(size_t)k * 128] : 0.f);
    }
    v8f c = {};
    c = wmma_bf16(a, b, c);
#pragma unroll
    for (int r = 0; r < 8; ++r) {
      const int row = r + ((lane >> 4) << 3);
      if (row < 9) atomicAdd(&node[(size_t)d * 1152 + row * 128 + n], c[r] * alpha);
    }
  }
}

// ---------------------------------------------------------------------------
// Kernel 9: out[n,k,:] = node[n,k,:] @ proj_w[ELL[k]].T (+ proj_b at k==0)
// ---------------------------------------------------------------------------
__launch_bounds__(256)
__global__ void k_proj(const float* __restrict__ node,
                       const bf16* __restrict__ proj_wb, const float* __restrict__ proj_b,
                       float* __restrict__ out) {
  const int n0   = blockIdx.x * 16;
  const int wave = threadIdx.x >> 5;
  const int lane = threadIdx.x & 31;
  for (int u = wave; u < 72; u += 8) {
    const int k = u >> 3, nt = u & 7;
    const int ell = (k == 0) ? 0 : (k < 4 ? 1 : 2);   // ELL = [0,1,1,1,2,2,2,2,2]
    const bf16* pw = proj_wb + (size_t)ell * 128 * 128;
    v8f c = {};
    for (int kb = 0; kb < 4; ++kb) {
      v16bf a;
      {
        const int m = lane & 15;
        const int row = n0 + m;
        const bool ok = row < kN;
        const float* p = node + (size_t)row * 1152 + k * 128 + kb * 32 + ((lane >> 4) << 4);
#pragma unroll
        for (int j = 0; j < 8; ++j) {
          a[j]     = (bf16)(ok ? p[j] : 0.f);
          a[8 + j] = (bf16)(ok ? p[8 + j] : 0.f);
        }
      }
      v16bf b = frag_b_bf16(pw, 128, nt * 16, kb * 32);
      c = wmma_bf16(a, b, c);
    }
    const int col = nt * 16 + (lane & 15);
    const float bias = (k == 0) ? proj_b[col] : 0.f;
#pragma unroll
    for (int r = 0; r < 8; ++r) {
      const int row = r + ((lane >> 4) << 3);
      const int n = n0 + row;
      if (n < kN) out[(size_t)n * 1152 + k * 128 + col] = c[r] + bias;
    }
  }
}

// ---------------------------------------------------------------------------
// Host launcher
// ---------------------------------------------------------------------------
extern "C" void kernel_launch(void* const* d_in, const int* in_sizes, int n_in,
                              void* d_out, int out_size, void* d_ws, size_t ws_size,
                              hipStream_t stream) {
  (void)in_sizes; (void)n_in; (void)out_size; (void)ws_size;
  const float* x          = (const float*)d_in[0];
  const int*   atomic_num = (const int*)  d_in[1];
  const float* edge_dist  = (const float*)d_in[2];
  const int*   edge_index = (const int*)  d_in[3];
  const float* wig        = (const float*)d_in[4];
  const float* wig_inv    = (const float*)d_in[5];
  const float* to_grid    = (const float*)d_in[6];
  const float* from_grid  = (const float*)d_in[7];
  const float* src_emb    = (const float*)d_in[8];
  const float* tgt_emb    = (const float*)d_in[9];
  const float* rad_w1     = (const float*)d_in[10];
  const float* rad_b1     = (const float*)d_in[11];
  const float* rad_g1     = (const float*)d_in[12];
  const float* rad_bb1    = (const float*)d_in[13];
  const float* rad_w2     = (const float*)d_in[14];
  const float* rad_b2     = (const float*)d_in[15];
  const float* rad_g2     = (const float*)d_in[16];
  const float* rad_bb2    = (const float*)d_in[17];
  const float* rad_w3     = (const float*)d_in[18];
  const float* rad_b3     = (const float*)d_in[19];
  const float* c1_w0      = (const float*)d_in[20];
  const float* c1_b0      = (const float*)d_in[21];
  const float* c1_w1      = (const float*)d_in[22];
  const float* c1_w2      = (const float*)d_in[23];
  const float* c2_w0      = (const float*)d_in[24];
  const float* c2_b0      = (const float*)d_in[25];
  const float* c2_w1      = (const float*)d_in[26];
  const float* c2_w2      = (const float*)d_in[27];
  const float* aln_g      = (const float*)d_in[28];
  const float* aln_b      = (const float*)d_in[29];
  const float* alpha_dot  = (const float*)d_in[30];
  const float* proj_w     = (const float*)d_in[31];
  const float* proj_b     = (const float*)d_in[32];

  float* ws = (float*)d_ws;
  size_t off = 0;
  float* rad    = ws + off; off += (size_t)kE * 768;    // radial features
  float* msgW   = ws + off; off += (size_t)kE * 1152;   // rotated msg / reused as msg2
  float* msg1   = ws + off; off += (size_t)kE * 576;    // conv1 output (E,9,64)
  float* extraD = ws + off; off += (size_t)kE * 576;    // alpha(512)+gating(64)
  float* eaL    = ws + off; off += (size_t)kE * 8;      // logits -> exp(logit-max)
  float* amax   = ws + off; off += (size_t)kN * 8;
  float* denom  = ws + off; off += (size_t)kN * 8;
  float* node   = ws + off; off += (size_t)kN * 1152;   // segment-sum accumulator
  float* msg2   = msgW;                                  // buffer reuse

  // bf16 weight arena after the fp32 region
  bf16* wsb = (bf16*)(ws + off);
  size_t boff = 0;
  bf16* rad_w1b = wsb + boff; boff += 64 * 256;
  bf16* rad_w2b = wsb + boff; boff += 64 * 64;
  bf16* rad_w3b = wsb + boff; boff += 768 * 64;
  bf16* c1_w0b  = wsb + boff; boff += 768 * 384;
  bf16* c1_w1f  = wsb + boff; boff += 256 * 512;   // folded (2h=256 x 2K=512)
  bf16* c1_w2f  = wsb + boff; boff += 128 * 256;   // folded (128 x 256)
  bf16* c2_w0b  = wsb + boff; boff += 384 * 192;
  bf16* c2_w1f  = wsb + boff; boff += 512 * 256;   // folded (512 x 256)
  bf16* c2_w2f  = wsb + boff; boff += 256 * 128;   // folded (256 x 128)
  bf16* proj_wb = wsb + boff; boff += 3 * 128 * 128;

  // ---- weight prep (bf16 transcode + complex fold) ----
  k_cvt<<<(64 * 256 + 255) / 256, 256, 0, stream>>>(rad_w1, rad_w1b, 64 * 256);
  k_cvt<<<(64 * 64 + 255) / 256, 256, 0, stream>>>(rad_w2, rad_w2b, 64 * 64);
  k_cvt<<<(768 * 64 + 255) / 256, 256, 0, stream>>>(rad_w3, rad_w3b, 768 * 64);
  k_cvt<<<(768 * 384 + 255) / 256, 256, 0, stream>>>(c1_w0, c1_w0b, 768 * 384);
  k_cvt<<<(384 * 192 + 255) / 256, 256, 0, stream>>>(c2_w0, c2_w0b, 384 * 192);
  k_cvt<<<(3 * 128 * 128 + 255) / 256, 256, 0, stream>>>(proj_w, proj_wb, 3 * 128 * 128);
  k_fold<<<(256 * 512 + 255) / 256, 256, 0, stream>>>(c1_w1, c1_w1f, 256, 128, 256 * 512);
  k_fold<<<(128 * 256 + 255) / 256, 256, 0, stream>>>(c1_w2, c1_w2f, 128, 64, 128 * 256);
  k_fold<<<(512 * 256 + 255) / 256, 256, 0, stream>>>(c2_w1, c2_w1f, 128, 256, 512 * 256);
  k_fold<<<(256 * 128 + 255) / 256, 256, 0, stream>>>(c2_w2, c2_w2f, 64, 128, 256 * 128);

  hipMemsetAsync(node, 0, (size_t)kN * 1152 * sizeof(float), stream);
  k_init<<<(kN * 8 + 255) / 256, 256, 0, stream>>>(amax, denom, kN * 8);

  // ---- pipeline ----
  k_rad_mlp<<<kE / 16, 128, 0, stream>>>(edge_dist, atomic_num, edge_index,
                                         src_emb, tgt_emb,
                                         rad_w1b, rad_b1, rad_g1, rad_bb1,
                                         rad_w2b, rad_b2, rad_g2, rad_bb2,
                                         rad_w3b, rad_b3, rad);
  k_wigner_msg<<<kE / 8, 256, 0, stream>>>(x, wig, edge_index, msgW);
  k_conv1<<<kE / 16, 256, 0, stream>>>(msgW, rad, c1_w0b, c1_b0, c1_w1f, c1_w2f,
                                       extraD, msg1);
  k_logits<<<(kE * 8 + 255) / 256, 256, 0, stream>>>(extraD, aln_g, aln_b,
                                                     alpha_dot, edge_index, eaL, amax);
  k_ea<<<(kE * 8 + 255) / 256, 256, 0, stream>>>(amax, edge_index, eaL, denom);
  k_grid<<<(kE * 64) / 256, 256, 0, stream>>>(to_grid, from_grid, extraD, msg1);
  k_conv2<<<kE / 16, 256, 0, stream>>>(msg1, c2_w0b, c2_b0, c2_w1f, c2_w2f, msg2);
  k_attn_agg<<<kE / 8, 256, 0, stream>>>(msg2, wig_inv, edge_index, eaL, denom, node);
  k_proj<<<(kN + 15) / 16, 256, 0, stream>>>(node, proj_wb, proj_b, (float*)d_out);
}
